// MultiHeadCrossAttention_45045617000764
// MI455X (gfx1250) — compile-verified, matrix operand order checked by reading, not run on hardware
//
#include <hip/hip_runtime.h>
#include <hip/hip_bf16.h>
#include <stdint.h>

// Problem constants (from reference)
#define B_   4
#define SQ_  2048
#define SK_  2048
#define HID_ 1024
#define NH_  16
#define HD_  64

typedef __attribute__((ext_vector_type(16))) __bf16 v16bf;
typedef __attribute__((ext_vector_type(8)))  float  v8f;
typedef __attribute__((ext_vector_type(4)))  int    v4i_t;

#define AS1 __attribute__((address_space(1)))
#define AS3 __attribute__((address_space(3)))

#if defined(__HIP_DEVICE_COMPILE__) && __has_builtin(__builtin_amdgcn_global_load_async_to_lds_b128)
#define HAVE_ASYNC_LDS 1
#else
#define HAVE_ASYNC_LDS 0
#endif

// ---------- helpers ----------
static __device__ __forceinline__ unsigned short f32_to_bf16(float f) {
  union { float f; uint32_t u; } c; c.f = f;
  uint32_t r = c.u + 0x7FFFu + ((c.u >> 16) & 1u);   // round-to-nearest-even
  return (unsigned short)(r >> 16);
}
static __device__ __forceinline__ uint32_t pack2_bf16(float x, float y) {
  return (uint32_t)f32_to_bf16(x) | ((uint32_t)f32_to_bf16(y) << 16);
}
static __device__ __forceinline__ void store_out(float* p, float v)          { *p = v; }
static __device__ __forceinline__ void store_out(unsigned short* p, float v) { *p = f32_to_bf16(v); }

#if HAVE_ASYNC_LDS
// 16-byte async DMA: global -> LDS (GLOBAL_LOAD_ASYNC_TO_LDS_B128, ASYNCcnt)
static __device__ __forceinline__ void async_copy16(const void* g, void* l) {
  __builtin_amdgcn_global_load_async_to_lds_b128(
      (AS1 v4i_t*)g, (AS3 v4i_t*)l, 0, 0);
}
#endif

static __device__ __forceinline__ void wait_async0() {
#if HAVE_ASYNC_LDS
  asm volatile("s_wait_asynccnt 0" ::: "memory");
#endif
}

union FragB16 { v16bf v; uint32_t d[8]; };

// Load a 16x32 bf16 WMMA fragment from an LDS tile stored [row][reduction]
// with reduction contiguous. Per ISA 16-bit A layout: lane<16 holds K in
// {0..7,16..23}, lane>=16 holds K in {8..15,24..31}; pairs packed per VGPR.
// Lowers to two ds_load_b128 per fragment.
static __device__ __forceinline__ v16bf load_frag(const unsigned short* base, int ld,
                                                  int row0, int red0, int lane) {
  FragB16 u;
  const unsigned short* rp = base + (row0 + (lane & 15)) * ld + red0 + ((lane >> 4) << 3);
#pragma unroll
  for (int j = 0; j < 8; ++j) {
    int kb = ((j & 4) << 2) + ((j & 3) << 1);   // j<4: 2j ; j>=4: 16+2(j-4)
    u.d[j] = *(const uint32_t*)(rp + kb);
  }
  return u.v;
}

static __device__ __forceinline__ v8f wmma_bf16(v16bf a, v16bf b, v8f c) {
  return __builtin_amdgcn_wmma_f32_16x16x32_bf16(
      false, a, false, b, (short)0, c, false, false);
}

// ---------- one-shot fp32 -> bf16 conversion (activations) ----------
// 16 elements per thread, 1024 float4 chunks per block.
__global__ __launch_bounds__(256) void cvt_f32_bf16(const float* __restrict__ src,
                                                    unsigned short* __restrict__ dst) {
#pragma unroll
  for (int i = 0; i < 4; ++i) {
    long c = ((long)blockIdx.x << 10) + (i << 8) + threadIdx.x;   // float4 chunk id
    const float4 v = *(const float4*)(src + (c << 2));
    uint2 p;
    p.x = pack2_bf16(v.x, v.y);
    p.y = pack2_bf16(v.z, v.w);
    *(uint2*)(dst + (c << 2)) = p;
  }
}

// ---------- one-shot W[K][N] fp32 -> Wt[N][K] bf16 (transpose+convert) ----------
__global__ __launch_bounds__(256) void transpose_cvt(const float* __restrict__ W,
                                                     unsigned short* __restrict__ Wt,
                                                     int K, int N) {
  __shared__ unsigned short T[32 * 36];        // [k][n] tile, padded
  const int tid = threadIdx.x;
  const int k0 = blockIdx.y << 5, n0 = blockIdx.x << 5;
  {
    int r = tid >> 3, c4 = (tid & 7) << 2;     // r: k-row, c4: n-col
    const float4 v = *(const float4*)(W + (long)(k0 + r) * N + n0 + c4);
    uint2 p;
    p.x = pack2_bf16(v.x, v.y);
    p.y = pack2_bf16(v.z, v.w);
    *(uint2*)&T[r * 36 + c4] = p;
  }
  __syncthreads();
  {
    int n = tid >> 3, k4 = (tid & 7) << 2;     // output row n, k chunk
    uint2 p;
    p.x = (uint32_t)T[(k4 + 0) * 36 + n] | ((uint32_t)T[(k4 + 1) * 36 + n] << 16);
    p.y = (uint32_t)T[(k4 + 2) * 36 + n] | ((uint32_t)T[(k4 + 3) * 36 + n] << 16);
    *(uint2*)(Wt + (long)(n0 + n) * K + k0 + k4) = p;
  }
}

// ---------- GEMM with bias: out = (A @ Wt^T + bias) * outScale ----------
// A: [M][K] bf16, Wt: [N][K] bf16 (pre-transposed), bias: [N] fp32.
// Block tile 128x128, BK=32, 256 threads (8 waves), wave 32x64 (2x4 WMMA).
// Double-buffered LDS; tiles arrive purely via async DMA (no VGPR staging).
template <typename OutT>
__global__ __launch_bounds__(256) void gemm_bias_wmma(
    const unsigned short* __restrict__ A, const unsigned short* __restrict__ Wt,
    const float* __restrict__ bias, OutT* __restrict__ out,
    int M, int N, int K, float outScale)
{
  constexpr int BM = 128, BN = 128, BK = 32, LDA = 40; // pitch 40: 16B rows, bank-clean
  __shared__ unsigned short As[2][BM * LDA];   // [m][k], k contiguous
  __shared__ unsigned short Bs[2][BN * LDA];   // [n][k], k contiguous

  const int tid  = threadIdx.x;
  const int w    = tid >> 5, lane = tid & 31;
  const int wm   = (w & 3) << 5;               // wave M offset: 0,32,64,96
  const int wn   = (w >> 2) << 6;              // wave N offset: 0 or 64
  const int m0   = blockIdx.y * BM;
  const int n0   = blockIdx.x * BN;
  const int l15  = lane & 15, half8 = (lane >> 4) << 3;

  v8f acc[2][4];
#pragma unroll
  for (int i = 0; i < 2; ++i)
#pragma unroll
    for (int j = 0; j < 4; ++j) acc[i][j] = v8f{};

  auto load_tiles = [&](int kt, int buf) {
#pragma unroll
    for (int i = 0; i < 2; ++i) {              // 512 b128 chunks per tile (A and B)
      int lin = tid + (i << 8);
      int r = lin >> 2, c8 = (lin & 3) << 3;
      const unsigned short* ga = A  + (long)(m0 + r) * K + kt + c8;
      const unsigned short* gb = Wt + (long)(n0 + r) * K + kt + c8;
#if HAVE_ASYNC_LDS
      async_copy16(ga, &As[buf][r * LDA + c8]);
      async_copy16(gb, &Bs[buf][r * LDA + c8]);
#else
      *(uint4*)&As[buf][r * LDA + c8] = *(const uint4*)ga;
      *(uint4*)&Bs[buf][r * LDA + c8] = *(const uint4*)gb;
#endif
    }
  };

  load_tiles(0, 0);
  wait_async0();
  __syncthreads();

  for (int kt = 0; kt < K; kt += BK) {
    const int cur = (kt >> 5) & 1;
    if (kt + BK < K) load_tiles(kt + BK, cur ^ 1);   // prefetch next tile (async)

#pragma unroll
    for (int i = 0; i < 2; ++i) {
      v16bf af = load_frag(As[cur], LDA, wm + (i << 4), 0, lane);
#pragma unroll
      for (int j = 0; j < 4; ++j) {
        v16bf bf = load_frag(Bs[cur], LDA, wn + (j << 4), 0, lane);
        acc[i][j] = wmma_bf16(af, bf, acc[i][j]);
      }
    }
    wait_async0();                                    // next tile DMA complete
    __syncthreads();                                  // all waves done with cur
  }

  float bcol[4];
#pragma unroll
  for (int j = 0; j < 4; ++j) bcol[j] = bias[n0 + wn + (j << 4) + l15];

#pragma unroll
  for (int i = 0; i < 2; ++i)
#pragma unroll
    for (int j = 0; j < 4; ++j)
#pragma unroll
      for (int v = 0; v < 8; ++v) {
        int row = m0 + wm + (i << 4) + v + half8;     // C layout: VGPR v -> M=v(+8)
        int col = n0 + wn + (j << 4) + l15;
        float val = (acc[i][j][v] + bcol[j]) * outScale;
        store_out(out + (long)row * N + col, val);
      }
}

// ---------- Flash attention (streaming softmax) ----------
// One block per (b, h, 64-row q-tile). Q was pre-scaled by HD^-0.5.
// Q,K,V,O are bf16 workspaces laid out [b*S + s][HID] with head offset h*HD.
__global__ __launch_bounds__(256) void flash_attn_wmma(
    const unsigned short* __restrict__ Q, const unsigned short* __restrict__ K,
    const unsigned short* __restrict__ V, unsigned short* __restrict__ O)
{
  constexpr int KP = 72;                     // bf16 tile pitch: 144B rows, bank-clean
  constexpr int SP = 68;                     // f32 tile pitch: 272B rows, bank-clean
  __shared__ unsigned short Qlds[64 * KP];   // [q][d]
  __shared__ unsigned short Klds[64 * KP];   // [k][d]   (B operand of Q*K^T)
  __shared__ unsigned short Vtlds[64 * KP];  // [d][k]   (B operand of P*V)
  __shared__ float          Slds[64 * SP];   // scores / exp
  __shared__ unsigned short Plds[64 * KP];   // exp as bf16
  __shared__ float m_lds[64], a_lds[64], l_lds[64];

  const int tid  = threadIdx.x;
  const int w    = tid >> 5, lane = tid & 31;
  const int l15  = lane & 15, half8 = (lane >> 4) << 3;
  const int b    = blockIdx.y >> 4;          // NH = 16
  const int h    = blockIdx.y & 15;
  const int q0   = blockIdx.x * 64;
  const long qbase = ((long)b * SQ_ + q0) * HID_ + h * HD_;

  // Load Q tile (64x64 bf16), b128 per lane
#pragma unroll
  for (int i = 0; i < 2; ++i) {
    int lin = tid + (i << 8);
    int r = lin >> 3, c8 = (lin & 7) << 3;
    const unsigned short* gp = Q + qbase + (long)r * HID_ + c8;
#if HAVE_ASYNC_LDS
    async_copy16(gp, &Qlds[r * KP + c8]);
#else
    *(uint4*)&Qlds[r * KP + c8] = *(const uint4*)gp;
#endif
  }

  const int qi = (w & 3) << 4;               // wave's q sub-row base (0..48)
  const int cj = (w >> 2) << 5;              // wave's column pair base (0/32)

  v8f oacc[2]; oacc[0] = v8f{}; oacc[1] = v8f{};
  float m_i = -1e30f, l_i = 0.f;             // row stats, owned by tid<64

  for (int kt = 0; kt < SK_ / 64; ++kt) {
    __syncthreads();                         // prev iteration consumers done
    const long kbase = ((long)b * SK_ + kt * 64) * HID_ + h * HD_;
#pragma unroll
    for (int i = 0; i < 2; ++i) {
      int lin = tid + (i << 8);
      int r = lin >> 3, c8 = (lin & 7) << 3;
      // K tile: straight copy -> async DMA when available
      const unsigned short* gk = K + kbase + (long)r * HID_ + c8;
#if HAVE_ASYNC_LDS
      async_copy16(gk, &Klds[r * KP + c8]);
#else
      *(uint4*)&Klds[r * KP + c8] = *(const uint4*)gk;
#endif
      // V tile: needs transpose -> b128 load, scalar LDS scatter
      uint4 vv = *(const uint4*)(V + kbase + (long)r * HID_ + c8);
      const unsigned short* pv = (const unsigned short*)&vv;
#pragma unroll
      for (int e = 0; e < 8; ++e) Vtlds[(c8 + e) * KP + r] = pv[e];
    }
    wait_async0();
    __syncthreads();

    // S = Q @ K^T  (scale already folded into Q)
#pragma unroll
    for (int t = 0; t < 2; ++t) {
      v8f s = v8f{};
#pragma unroll
      for (int dd = 0; dd < 64; dd += 32) {
        v16bf af = load_frag(Qlds, KP, qi, dd, lane);
        v16bf bf = load_frag(Klds, KP, cj + (t << 4), dd, lane);
        s = wmma_bf16(af, bf, s);
      }
#pragma unroll
      for (int v = 0; v < 8; ++v)
        Slds[(qi + v + half8) * SP + cj + (t << 4) + l15] = s[v];
    }
    __syncthreads();

    // Online softmax stats (one thread per row, float4)
    if (tid < 64) {
      const float4* rp = (const float4*)&Slds[tid * SP];
      float rmax = -1e30f;
#pragma unroll
      for (int c = 0; c < 16; ++c) {
        float4 v = rp[c];
        rmax = fmaxf(rmax, fmaxf(fmaxf(v.x, v.y), fmaxf(v.z, v.w)));
      }
      float mn = fmaxf(m_i, rmax);
      a_lds[tid] = __expf(m_i - mn);
      m_lds[tid] = mn;
      m_i = mn;
    }
    __syncthreads();

    // Exponentiate (float4); keep f32 in Slds (row sums), bf16 in Plds (WMMA)
#pragma unroll
    for (int i = 0; i < 4; ++i) {
      int lin = tid + (i << 8);                // 1024 float4 chunks
      int r = lin >> 4, c = (lin & 15) << 2;
      float4 s4 = *(float4*)&Slds[r * SP + c];
      float mr = m_lds[r];
      s4.x = __expf(s4.x - mr); s4.y = __expf(s4.y - mr);
      s4.z = __expf(s4.z - mr); s4.w = __expf(s4.w - mr);
      *(float4*)&Slds[r * SP + c] = s4;
      uint2 p;
      p.x = pack2_bf16(s4.x, s4.y);
      p.y = pack2_bf16(s4.z, s4.w);
      *(uint2*)&Plds[r * KP + c] = p;
    }
    __syncthreads();

    // Rescale O accumulators by alpha
#pragma unroll
    for (int t = 0; t < 2; ++t)
#pragma unroll
      for (int v = 0; v < 8; ++v)
        oacc[t][v] *= a_lds[qi + v + half8];

    // Update row sums (float4)
    if (tid < 64) {
      const float4* rp = (const float4*)&Slds[tid * SP];
      float s = 0.f;
#pragma unroll
      for (int c = 0; c < 16; ++c) {
        float4 v = rp[c];
        s += (v.x + v.y) + (v.z + v.w);
      }
      l_i = l_i * a_lds[tid] + s;
    }

    // O += P @ V
#pragma unroll
    for (int t = 0; t < 2; ++t) {
#pragma unroll
      for (int kk = 0; kk < 64; kk += 32) {
        v16bf af = load_frag(Plds, KP, qi, kk, lane);
        v16bf bf = load_frag(Vtlds, KP, cj + (t << 4), kk, lane);
        oacc[t] = wmma_bf16(af, bf, oacc[t]);
      }
    }
  }

  __syncthreads();
  if (tid < 64) l_lds[tid] = l_i;
  __syncthreads();

#pragma unroll
  for (int t = 0; t < 2; ++t)
#pragma unroll
    for (int v = 0; v < 8; ++v) {
      int r    = qi + v + half8;
      int dcol = cj + (t << 4) + l15;
      float val = oacc[t][v] / l_lds[r];
      O[qbase + (long)r * HID_ + dcol] = f32_to_bf16(val);
    }
}

// ---------- launch ----------
extern "C" void kernel_launch(void* const* d_in, const int* in_sizes, int n_in,
                              void* d_out, int out_size, void* d_ws, size_t ws_size,
                              hipStream_t stream) {
  (void)in_sizes; (void)n_in; (void)out_size; (void)ws_size;
  const float* query = (const float*)d_in[0];
  const float* key   = (const float*)d_in[1];
  const float* value = (const float*)d_in[2];
  const float* Wq = (const float*)d_in[3];
  const float* bq = (const float*)d_in[4];
  const float* Wk = (const float*)d_in[5];
  const float* bk = (const float*)d_in[6];
  const float* Wv = (const float*)d_in[7];
  const float* bv = (const float*)d_in[8];
  const float* Wo = (const float*)d_in[9];
  const float* bo = (const float*)d_in[10];
  float* out = (float*)d_out;

  const int    M  = B_ * SQ_;                  // 8192
  const size_t ME = (size_t)M * HID_;          // 8.4M elems (16 MB bf16)
  // Workspace layout (bf16 elems): Qws | Kws | Vws | X(=Ows) | Wt  => 66 MB total
  unsigned short* Qws = (unsigned short*)d_ws;
  unsigned short* Kws = Qws + ME;
  unsigned short* Vws = Kws + ME;
  unsigned short* X   = Vws + ME;              // staged bf16 activations; later Ows
  unsigned short* Ows = X;
  unsigned short* Wt  = X + ME;                // 1024x1024 bf16, reused per GEMM

  dim3 blk(256);
  dim3 cgrid(ME / 4096);                       // cvt: 1024 float4 chunks / block
  dim3 tgrid(HID_ / 32, HID_ / 32);            // transpose: 32x32 tiles
  dim3 ggrid(HID_ / 128, M / 128);             // gemm: 128x128 tiles
  dim3 agrid(SQ_ / 64, B_ * NH_);              // attention
  const float qscale = 0.125f;                 // HD^-0.5 = 1/sqrt(64)

  // Q projection (scale folded into output)
  cvt_f32_bf16<<<cgrid, blk, 0, stream>>>(query, X);
  transpose_cvt<<<tgrid, blk, 0, stream>>>(Wq, Wt, HID_, HID_);
  gemm_bias_wmma<unsigned short><<<ggrid, blk, 0, stream>>>(
      X, Wt, bq, Qws, M, HID_, HID_, qscale);
  // K projection
  cvt_f32_bf16<<<cgrid, blk, 0, stream>>>(key, X);
  transpose_cvt<<<tgrid, blk, 0, stream>>>(Wk, Wt, HID_, HID_);
  gemm_bias_wmma<unsigned short><<<ggrid, blk, 0, stream>>>(
      X, Wt, bk, Kws, M, HID_, HID_, 1.0f);
  // V projection
  cvt_f32_bf16<<<cgrid, blk, 0, stream>>>(value, X);
  transpose_cvt<<<tgrid, blk, 0, stream>>>(Wv, Wt, HID_, HID_);
  gemm_bias_wmma<unsigned short><<<ggrid, blk, 0, stream>>>(
      X, Wt, bv, Vws, M, HID_, HID_, 1.0f);

  // Attention (writes Ows, aliasing the now-dead X buffer)
  flash_attn_wmma<<<agrid, blk, 0, stream>>>(Qws, Kws, Vws, Ows);

  // Output projection -> fp32 d_out
  transpose_cvt<<<tgrid, blk, 0, stream>>>(Wo, Wt, HID_, HID_);
  gemm_bias_wmma<float><<<ggrid, blk, 0, stream>>>(
      Ows, Wt, bo, out, M, HID_, HID_, 1.0f);
}
